// EdgeModel_17497696764456
// MI455X (gfx1250) — matrix operand order repaired and checked by reading, not compiled, not run on hardware
//
#include <hip/hip_runtime.h>

// ---------------------------------------------------------------------------
// EdgeModel MLP for MI455X (gfx1250, wave32, WMMA bf16)
//   x = concat[ek(64) | vrk(128) | vsk(128) | u[batch](64)]   -> [E,384]
//   h = relu(x @ W1 + b1)   W1:[384,128]
//   o = h @ W2 + b2         W2:[128, 64]
// ---------------------------------------------------------------------------

typedef __attribute__((ext_vector_type(16))) __bf16 v16bf;
typedef __attribute__((ext_vector_type(8)))  float  v8f;

#define E_EDGES 320000
#define NNF 128
#define NEF 64
#define NGF 64
#define HID 128
#define TGT 64
#define IN_CH 384

// ws layout (dwords): W1 fragments [12 kt][8 nt][32 lane][8 dw]  = 24576 dwords
//                     W2 fragments [ 4 kt][4 nt][32 lane][8 dw]  =  4096 dwords
#define W1F_DW 24576
#define W2F_DW 4096

static __device__ __forceinline__ unsigned pack_bf16(float a, float b) {
    unsigned lo = (unsigned)__builtin_bit_cast(unsigned short, (__bf16)a);
    unsigned hi = (unsigned)__builtin_bit_cast(unsigned short, (__bf16)b);
    return lo | (hi << 16);
}

// Pre-swizzle W1/W2 (f32 row-major [K][N]) into per-lane WMMA B-fragment
// layout: lane = g*16+n holds column N = nt*16+n, dword d holds K pair
// (kt*32 + 8g + 2d + (d>=4 ? 8 : 0), +1).  One contiguous 32B chunk per lane per fragment.
__global__ void prep_weights(const float* __restrict__ W1,
                             const float* __restrict__ W2,
                             unsigned* __restrict__ wf) {
    int t = blockIdx.x * blockDim.x + threadIdx.x;
    if (t < W1F_DW) {
        int d = t & 7, lane = (t >> 3) & 31, frag = t >> 8;
        int nt = frag & 7, kt = frag >> 3;
        int K = kt * 32 + ((lane >> 4) << 3) + 2 * d + ((d >> 2) << 3);
        int N = nt * 16 + (lane & 15);
        wf[t] = pack_bf16(W1[K * HID + N], W1[(K + 1) * HID + N]);
    } else if (t < W1F_DW + W2F_DW) {
        int s = t - W1F_DW;
        int d = s & 7, lane = (s >> 3) & 31, frag = s >> 8;
        int nt = frag & 3, kt = frag >> 2;
        int K = kt * 32 + ((lane >> 4) << 3) + 2 * d + ((d >> 2) << 3);
        int N = nt * 16 + (lane & 15);
        wf[t] = pack_bf16(W2[K * TGT + N], W2[(K + 1) * TGT + N]);
    }
}

static __device__ __forceinline__ v16bf cvt_a_frag(float4 c0, float4 c1,
                                                   float4 c2, float4 c3) {
    v16bf a;
    a[0]  = (__bf16)c0.x; a[1]  = (__bf16)c0.y; a[2]  = (__bf16)c0.z; a[3]  = (__bf16)c0.w;
    a[4]  = (__bf16)c1.x; a[5]  = (__bf16)c1.y; a[6]  = (__bf16)c1.z; a[7]  = (__bf16)c1.w;
    a[8]  = (__bf16)c2.x; a[9]  = (__bf16)c2.y; a[10] = (__bf16)c2.z; a[11] = (__bf16)c2.w;
    a[12] = (__bf16)c3.x; a[13] = (__bf16)c3.y; a[14] = (__bf16)c3.z; a[15] = (__bf16)c3.w;
    return a;
}

#define HSTRIDE 144  // 128 + 16 bf16 pad: keeps 16B alignment, spreads banks

__global__ __launch_bounds__(256, 1)
void edge_mlp(const float* __restrict__ ek,  const float* __restrict__ vrk,
              const float* __restrict__ vsk, const float* __restrict__ u,
              const int*   __restrict__ batch,
              const float* __restrict__ b1,  const float* __restrict__ b2,
              const v16bf* __restrict__ w1f, const v16bf* __restrict__ w2f,
              float* __restrict__ out) {
    // per-wave h-staging buffer: 32 rows x HSTRIDE bf16
    __shared__ __bf16 hbuf[8 * 32 * HSTRIDE];

    const int lane = threadIdx.x & 31;
    const int wave = threadIdx.x >> 5;
    const int tile = blockIdx.x * 8 + wave;        // 32 edges per wave
    if (tile * 32 >= E_EDGES) return;              // uniform per wave

    const int ln = lane & 15;
    const int g  = lane >> 4;
    const int rowBase = tile * 32;
    const int r0 = rowBase + ln;                   // M-subtile 0 row of this lane
    const int r1 = r0 + 16;                        // M-subtile 1 row
    const int bi0 = batch[r0];
    const int bi1 = batch[r1];

    __bf16* hw = hbuf + wave * 32 * HSTRIDE;

    const v8f zero8 = {0.f, 0.f, 0.f, 0.f, 0.f, 0.f, 0.f, 0.f};

    // ---------------- Layer 1: [32 x 384] x [384 x 128] ----------------
    v8f acc[2][8];
#pragma unroll
    for (int ms = 0; ms < 2; ++ms)
#pragma unroll
        for (int nt = 0; nt < 8; ++nt) acc[ms][nt] = zero8;

    for (int kt = 0; kt < 12; ++kt) {
        const int k0 = kt * 32;
        // region containing K block [k0, k0+32): uniform per kt
        const float* base; int stride, koff; bool isU = false;
        if (k0 < 64)       { base = ek;  stride = NEF; koff = k0;       }
        else if (k0 < 192) { base = vrk; stride = NNF; koff = k0 - 64;  }
        else if (k0 < 320) { base = vsk; stride = NNF; koff = k0 - 192; }
        else               { base = u;   stride = NGF; koff = k0 - 320; isU = true; }

        v16bf afr[2];
#pragma unroll
        for (int ms = 0; ms < 2; ++ms) {
            const int row = isU ? (ms ? bi1 : bi0) : (ms ? r1 : r0);
            const float* p = base + (long)row * stride + koff + 8 * g;
            float4 c0 = *(const float4*)(p);        // K = k0 + 8g + 0..3
            float4 c1 = *(const float4*)(p + 4);    // K = k0 + 8g + 4..7
            float4 c2 = *(const float4*)(p + 16);   // K = k0 + 16 + 8g + 0..3
            float4 c3 = *(const float4*)(p + 20);   // K = k0 + 16 + 8g + 4..7
            afr[ms] = cvt_a_frag(c0, c1, c2, c3);
        }
#pragma unroll
        for (int nt = 0; nt < 8; ++nt) {
            v16bf bf = w1f[(kt * 8 + nt) * 32 + lane];
            acc[0][nt] = __builtin_amdgcn_wmma_f32_16x16x32_bf16(
                false, afr[0], false, bf, (short)0, acc[0][nt], false, false);
            acc[1][nt] = __builtin_amdgcn_wmma_f32_16x16x32_bf16(
                false, afr[1], false, bf, (short)0, acc[1][nt], false, false);
        }
    }

    // bias + ReLU, transpose through LDS (D layout -> A layout)
#pragma unroll
    for (int nt = 0; nt < 8; ++nt) {
        const float bias = b1[nt * 16 + ln];
#pragma unroll
        for (int ms = 0; ms < 2; ++ms)
#pragma unroll
            for (int v = 0; v < 8; ++v) {
                float h = acc[ms][nt][v] + bias;
                h = fmaxf(h, 0.f);
                const int m = ms * 16 + v + 8 * g;        // row within tile
                hw[m * HSTRIDE + nt * 16 + ln] = (__bf16)h;
            }
    }
    __syncthreads();  // cross-lane LDS dependency (transpose)

    // ---------------- Layer 2: [32 x 128] x [128 x 64] ----------------
    v8f acc2[2][4];
#pragma unroll
    for (int ms = 0; ms < 2; ++ms)
#pragma unroll
        for (int nt = 0; nt < 4; ++nt) acc2[ms][nt] = zero8;

#pragma unroll
    for (int kt = 0; kt < 4; ++kt) {
        const int k0 = kt * 32;
        v16bf afr[2];
#pragma unroll
        for (int ms = 0; ms < 2; ++ms) {
            const __bf16* hp = hw + (ms * 16 + ln) * HSTRIDE + k0 + 8 * g;
            union { uint4 q[2]; v16bf v; } uu;
            uu.q[0] = *(const uint4*)(hp);        // K = k0 + 8g + 0..7
            uu.q[1] = *(const uint4*)(hp + 16);   // K = k0 + 16 + 8g + 0..7
            afr[ms] = uu.v;
        }
#pragma unroll
        for (int nt = 0; nt < 4; ++nt) {
            v16bf bf = w2f[(kt * 4 + nt) * 32 + lane];
            acc2[0][nt] = __builtin_amdgcn_wmma_f32_16x16x32_bf16(
                false, afr[0], false, bf, (short)0, acc2[0][nt], false, false);
            acc2[1][nt] = __builtin_amdgcn_wmma_f32_16x16x32_bf16(
                false, afr[1], false, bf, (short)0, acc2[1][nt], false, false);
        }
    }

    // ---------------- Output: bias + store [E x 64] f32 ----------------
#pragma unroll
    for (int nt = 0; nt < 4; ++nt) {
        const float bias = b2[nt * 16 + ln];
#pragma unroll
        for (int ms = 0; ms < 2; ++ms)
#pragma unroll
            for (int v = 0; v < 8; ++v) {
                const int row = rowBase + ms * 16 + v + 8 * g;
                out[(long)row * TGT + nt * 16 + ln] = acc2[ms][nt][v] + bias;
            }
    }
}

extern "C" void kernel_launch(void* const* d_in, const int* in_sizes, int n_in,
                              void* d_out, int out_size, void* d_ws, size_t ws_size,
                              hipStream_t stream) {
    const float* ek    = (const float*)d_in[0];
    const float* vrk   = (const float*)d_in[1];
    const float* vsk   = (const float*)d_in[2];
    const float* u     = (const float*)d_in[3];
    const int*   batch = (const int*)  d_in[4];
    const float* W1    = (const float*)d_in[5];
    const float* b1    = (const float*)d_in[6];
    const float* W2    = (const float*)d_in[7];
    const float* b2    = (const float*)d_in[8];
    float*    out = (float*)d_out;
    unsigned* wf  = (unsigned*)d_ws;

    // 1) swizzle weights into bf16 WMMA B-fragment layout (L2-resident)
    prep_weights<<<(W1F_DW + W2F_DW + 255) / 256, 256, 0, stream>>>(W1, W2, wf);

    // 2) fused gather + 2-layer MLP; 32 edges per wave, 8 waves per block
    const int tiles = E_EDGES / 32;  // 10000
    edge_mlp<<<tiles / 8, 256, 0, stream>>>(
        ek, vrk, vsk, u, batch, b1, b2,
        (const v16bf*)wf, (const v16bf*)(wf + W1F_DW), out);
}
